// MultiHeadAttention_35854386987288
// MI455X (gfx1250) — matrix operand order repaired and checked by reading, not compiled, not run on hardware
//
#include <hip/hip_runtime.h>
#include <hip/hip_bf16.h>

// ---------------------------------------------------------------------------
// MultiHeadAttention for MI455X (gfx1250, wave32, WMMA + TDM async staging).
// ---------------------------------------------------------------------------

typedef __attribute__((ext_vector_type(16))) _Float16 v16h;
typedef __attribute__((ext_vector_type(8)))  _Float16 v8h;
typedef __attribute__((ext_vector_type(8)))  float    v8f;

#define WMMA_F16(a, b, c) \
  __builtin_amdgcn_wmma_f32_16x16x32_f16(false, (a), false, (b), (short)0, (c), false, false)

#if defined(__has_builtin)
#if __has_builtin(__builtin_amdgcn_tensor_load_to_lds) && __has_builtin(__builtin_amdgcn_s_wait_tensorcnt)
#define HAVE_TDM 1
#endif
#endif
#ifndef HAVE_TDM
#define HAVE_TDM 0
#endif

static constexpr int Bc  = 4;
static constexpr int Sc  = 2048;
static constexpr int Hc  = 512;
static constexpr int NHc = 8;
static constexpr int DHc = 64;                 // head dim
static constexpr int Mrows = Bc * Sc;          // 8192
static constexpr float SCALE = 0.125f;         // 1/sqrt(64)

// LDS row strides (halves) for staged tiles: multiples of 16B, odd dword*4 banks
static constexpr int KT_STRIDE = 72;           // 64 data + 8 pad  -> 144 B rows
static constexpr int VT_STRIDE = 40;           // 32 data + 8 pad  ->  80 B rows

// ---- workspace layout (element offsets, _Float16) -------------------------
static constexpr size_t OFF_XF16 = 0;                               // [8192,512]
static constexpr size_t OFF_WQKV = OFF_XF16 + (size_t)Mrows * Hc;   // [1536,512]
static constexpr size_t OFF_OW   = OFF_WQKV + (size_t)3 * Hc * Hc;  // [512,512]
static constexpr size_t OFF_Q    = OFF_OW   + (size_t)Hc * Hc;      // [B,NH,S,DH]
static constexpr size_t OFF_K    = OFF_Q    + (size_t)Bc * NHc * Sc * DHc;
static constexpr size_t OFF_VT   = OFF_K    + (size_t)Bc * NHc * Sc * DHc;   // [B,NH,DH,S]
static constexpr size_t OFF_VALS = OFF_VT   + (size_t)Bc * NHc * Sc * DHc;   // [8192,512]

// ---------------------------------------------------------------------------
// A-fragment (16x32 f16, row-major, K contiguous).  lane L: row = L&15;
// halves[0:16) = K 16*(L>>4).. (contiguous)
// ---------------------------------------------------------------------------
__device__ __forceinline__ v16h load_a_frag(const _Float16* __restrict__ p, int ld) {
  const int lane = threadIdx.x & 31;
  const int lh = lane >> 4, r = lane & 15;
  const _Float16* row = p + (size_t)r * ld;
  v8h c0 = *(const v8h*)(row + 16 * lh);
  v8h c1 = *(const v8h*)(row + 16 * lh + 8);
  v16h a;
#pragma unroll
  for (int i = 0; i < 8; ++i) { a[i] = c0[i]; a[8 + i] = c1[i]; }
  return a;
}

// B-fragment (32x16) from matrix stored [N, K] row-major (compute A @ W^T):
// lane L holds column n = L&15, 16 contiguous K halves at 16*(L>>4).
__device__ __forceinline__ v16h load_bt_frag(const _Float16* __restrict__ p, int ld) {
  const int lane = threadIdx.x & 31;
  const int lh = lane >> 4, n = lane & 15;
  return *(const v16h*)(p + (size_t)n * ld + 16 * lh);
}

// Same B-fragment but from an LDS-staged tile with padded row stride.
__device__ __forceinline__ v16h lds_bfrag(const _Float16* tile, int strideHalves,
                                          int rowOff, int halfOff) {
  const int lane = threadIdx.x & 31;
  const _Float16* p = tile + (size_t)(rowOff + (lane & 15)) * strideHalves
                           + halfOff + 16 * (lane >> 4);
  v8h c0 = *(const v8h*)p;
  v8h c1 = *(const v8h*)(p + 8);
  v16h r;
#pragma unroll
  for (int i = 0; i < 8; ++i) { r[i] = c0[i]; r[8 + i] = c1[i]; }
  return r;
}

// ---------------------------------------------------------------------------
// TDM 2-D tile load (global -> LDS) with hardware row padding.
// Descriptor per CDNA5 ISA ch.8: group0 {count=1, lds_addr, global_addr, type=2},
// group1 {data_size=2B, pad_enable, pad_interval(p: 8<<p bytes), pad_amount(a: (a+1) dw),
//         tensor_dim0/1, tile_dim0/1, dim0_stride}.
// ---------------------------------------------------------------------------
#if HAVE_TDM
typedef __attribute__((ext_vector_type(4))) unsigned int v4u;
typedef __attribute__((ext_vector_type(4))) int v4i;
typedef __attribute__((ext_vector_type(8))) int v8i;

__device__ __forceinline__ void tdm_load_2d(const void* gsrc, void* ldst,
                                            int dim0, int dim1, int stride0,
                                            int tile0, int tile1,
                                            int padP, int padA) {
  unsigned long long ga = (unsigned long long)(uintptr_t)gsrc;
  unsigned galo = __builtin_amdgcn_readfirstlane((unsigned)ga);
  unsigned gahi = __builtin_amdgcn_readfirstlane((unsigned)(ga >> 32));
  unsigned ldsa = __builtin_amdgcn_readfirstlane((unsigned)(uintptr_t)ldst);
  v4u g0;
  g0[0] = 1u;                                      // count=1, user descriptor
  g0[1] = ldsa;                                    // lds_addr (bytes)
  g0[2] = galo;                                    // global_addr[31:0]
  g0[3] = (gahi & 0x01FFFFFFu) | (2u << 30);       // global_addr[56:32], type=2
  v8i g1;
  g1[0] = (1 << 16) | (1 << 20) | (padP << 22) | (padA << 25); // 2B elems, pad
  g1[1] = (dim0 & 0xFFFF) << 16;                   // tensor_dim0[15:0]
  g1[2] = ((unsigned)dim0 >> 16) | ((dim1 & 0xFFFF) << 16);
  g1[3] = ((unsigned)dim1 >> 16) | ((tile0 & 0xFFFF) << 16);
  g1[4] = (tile1 & 0xFFFF);                        // tile_dim1 (tile_dim2=0)
  g1[5] = stride0;                                 // dim0_stride[31:0]
  g1[6] = 0;
  g1[7] = 0;
  v4i z4 = (v4i){0, 0, 0, 0};
#if defined(__clang_major__) && __clang_major__ >= 23
  v8i z8 = (v8i){0, 0, 0, 0, 0, 0, 0, 0};
  __builtin_amdgcn_tensor_load_to_lds(g0, g1, z4, z4, z8, 0);
#else
  __builtin_amdgcn_tensor_load_to_lds(g0, g1, z4, z4, 0);
#endif
}
#endif  // HAVE_TDM

// ---------------------------------------------------------------------------
// Kernel 0: f32 -> f16 conversion
// ---------------------------------------------------------------------------
__global__ void mha_cvt_f16(const float* __restrict__ in, _Float16* __restrict__ out, int n) {
  int i = blockIdx.x * blockDim.x + threadIdx.x;
  if (i < n) out[i] = (_Float16)in[i];
}

// ---------------------------------------------------------------------------
// Kernel 1: QKV projection, software-pipelined WMMA GEMM, scatter into
// Q [b,h,s,d], K [b,h,s,d], Vt [b,h,d,s] (f16). Wave = 16x64 tile.
// ---------------------------------------------------------------------------
__global__ void mha_qkv_gemm(const _Float16* __restrict__ xf,
                             const _Float16* __restrict__ wq,
                             const float*    __restrict__ qkv_b,
                             _Float16* __restrict__ Q,
                             _Float16* __restrict__ K,
                             _Float16* __restrict__ Vt) {
  const int wave = threadIdx.x >> 5;
  const int lane = threadIdx.x & 31;
  const int lh = lane >> 4, lc = lane & 15;
  const int m0 = blockIdx.x * 16;
  const int n0 = (blockIdx.y * 4 + wave) * 64;    // 0..1472

  const _Float16* arow = xf + (size_t)m0 * Hc;

  v8f acc[4];
#pragma unroll
  for (int t = 0; t < 4; ++t) acc[t] = (v8f){};

  // software pipeline: fragments for step k+1 issued before WMMAs of step k
  v16h a = load_a_frag(arow, Hc);
  v16h b[4];
#pragma unroll
  for (int t = 0; t < 4; ++t) b[t] = load_bt_frag(wq + (size_t)(n0 + 16 * t) * Hc, Hc);

  for (int kk = 0; kk < Hc; kk += 32) {
    v16h an = a, bn0 = b[0], bn1 = b[1], bn2 = b[2], bn3 = b[3];
    if (kk + 32 < Hc) {
      __builtin_prefetch(arow + kk + 64, 0, 1);
      an  = load_a_frag(arow + kk + 32, Hc);
      bn0 = load_bt_frag(wq + (size_t)(n0 +  0) * Hc + kk + 32, Hc);
      bn1 = load_bt_frag(wq + (size_t)(n0 + 16) * Hc + kk + 32, Hc);
      bn2 = load_bt_frag(wq + (size_t)(n0 + 32) * Hc + kk + 32, Hc);
      bn3 = load_bt_frag(wq + (size_t)(n0 + 48) * Hc + kk + 32, Hc);
    }
    acc[0] = WMMA_F16(a, b[0], acc[0]);
    acc[1] = WMMA_F16(a, b[1], acc[1]);
    acc[2] = WMMA_F16(a, b[2], acc[2]);
    acc[3] = WMMA_F16(a, b[3], acc[3]);
    a = an; b[0] = bn0; b[1] = bn1; b[2] = bn2; b[3] = bn3;
  }

#pragma unroll
  for (int t = 0; t < 4; ++t) {
    const int n = n0 + 16 * t + lc;
    const int head = n / (3 * DHc);
    const int r = n - head * 3 * DHc;
    const float bias = qkv_b[n];
#pragma unroll
    for (int j = 0; j < 8; ++j) {
      const int m = m0 + j + 8 * lh;
      const int bb = m >> 11;
      const int s  = m & 2047;
      const _Float16 v = (_Float16)(acc[t][j] + bias);
      const size_t bh = (size_t)bb * NHc + head;
      if (r < DHc) {
        Q[(bh * Sc + s) * DHc + r] = v;                  // chunk0 = Q
      } else if (r < 2 * DHc) {
        Vt[(bh * DHc + (r - DHc)) * Sc + s] = v;         // chunk1 = V (transposed)
      } else {
        K[(bh * Sc + s) * DHc + (r - 2 * DHc)] = v;      // chunk2 = K
      }
    }
  }
}

// ---------------------------------------------------------------------------
// Kernel 2: flash attention. One wave per (b, head, 16-query tile).
// K/V tiles double-buffered in LDS, staged by TDM (async, TENSORcnt) when
// available. 32 keys/iter: 4 WMMA logits + online softmax + 4 WMMA P.V.
// ---------------------------------------------------------------------------
struct __align__(16) WaveLds {
  _Float16 kt[2][32 * KT_STRIDE];   // K tile [key][d], padded rows
  _Float16 vt[2][64 * VT_STRIDE];   // V tile [d][key], padded rows
  _Float16 p[16 * 40];              // P staging
};

__device__ __forceinline__ void stage_tile(const _Float16* __restrict__ Kh,
                                           const _Float16* __restrict__ Vth,
                                           int k0, WaveLds& L, int buf) {
#if HAVE_TDM
  // K: x = d (64 contiguous), y = 32 keys, row pad 32dw->+4dw (stride 144 B)
  tdm_load_2d(Kh + (size_t)k0 * DHc, L.kt[buf], DHc, Sc, DHc, DHc, 32, 4, 3);
  // V: x = 32 keys (contiguous in Vt), y = 64 d-rows, pad 16dw->+4dw (80 B)
  tdm_load_2d(Vth + k0, L.vt[buf], Sc, DHc, Sc, 32, DHc, 3, 3);
  __asm__ volatile("" ::: "memory");
#else
  const int lane = threadIdx.x & 31;
#pragma unroll
  for (int i = 0; i < 8; ++i) {
    const int ch = lane + 32 * i;                       // 256 x 16B chunks each
    { const int r = ch >> 3, c = ch & 7;                // K: 32 rows x 8 chunks
      *(v8h*)(L.kt[buf] + (size_t)r * KT_STRIDE + c * 8) =
          *(const v8h*)(Kh + (size_t)(k0 + r) * DHc + c * 8); }
    { const int r = ch >> 2, c = ch & 3;                // V: 64 rows x 4 chunks
      *(v8h*)(L.vt[buf] + (size_t)r * VT_STRIDE + c * 8) =
          *(const v8h*)(Vth + (size_t)r * Sc + k0 + c * 8); }
  }
  __builtin_amdgcn_wave_barrier();
#endif
}

__global__ void mha_flash_attn(const _Float16* __restrict__ Q,
                               const _Float16* __restrict__ K,
                               const _Float16* __restrict__ Vt,
                               const unsigned char* __restrict__ mask,  // [B,S] bool
                               _Float16* __restrict__ vals) {           // [B*S, H]
  __shared__ WaveLds lds[2];                 // 2 waves / block

  const int wave = threadIdx.x >> 5;
  const int lane = threadIdx.x & 31;
  const int lh = lane >> 4, lc = lane & 15;

  const int job  = blockIdx.x * 2 + wave;    // 0..4095
  const int qt   = job & 127;
  const int head = (job >> 7) & 7;
  const int bb   = job >> 10;
  const int q0   = qt * 16;

  const size_t bh = (size_t)bb * NHc + head;
  const _Float16* Qh  = Q  + bh * Sc * DHc;
  const _Float16* Kh  = K  + bh * Sc * DHc;
  const _Float16* Vth = Vt + bh * DHc * Sc;
  const unsigned char* mb = mask + (size_t)bb * Sc;
  WaveLds& L = lds[wave];

  const v16h aq0 = load_a_frag(Qh + (size_t)q0 * DHc + 0, DHc);
  const v16h aq1 = load_a_frag(Qh + (size_t)q0 * DHc + 32, DHc);

  v8f accv[4];
#pragma unroll
  for (int t = 0; t < 4; ++t) accv[t] = (v8f){};
  float mrow[8], lrow[8];
#pragma unroll
  for (int j = 0; j < 8; ++j) { mrow[j] = -1.0e30f; lrow[j] = 0.0f; }

  stage_tile(Kh, Vth, 0, L, 0);              // prime the pipeline

  for (int it = 0; it < Sc / 32; ++it) {
    const int k0  = it * 32;
    const int cur = it & 1;

    if (it + 1 < Sc / 32) {
      stage_tile(Kh, Vth, k0 + 32, L, cur ^ 1);
#if HAVE_TDM
      __builtin_amdgcn_s_wait_tensorcnt(2);  // current tile's pair retired
      __asm__ volatile("" ::: "memory");
#endif
    } else {
#if HAVE_TDM
      __builtin_amdgcn_s_wait_tensorcnt(0);
      __asm__ volatile("" ::: "memory");
#endif
    }

    // ---- logits: two 16-key tiles, K-dim = DH = 64
    v8f s0 = (v8f){}, s1 = (v8f){};
    {
      v16h bk;
      bk = lds_bfrag(L.kt[cur], KT_STRIDE,  0,  0); s0 = WMMA_F16(aq0, bk, s0);
      bk = lds_bfrag(L.kt[cur], KT_STRIDE,  0, 32); s0 = WMMA_F16(aq1, bk, s0);
      bk = lds_bfrag(L.kt[cur], KT_STRIDE, 16,  0); s1 = WMMA_F16(aq0, bk, s1);
      bk = lds_bfrag(L.kt[cur], KT_STRIDE, 16, 32); s1 = WMMA_F16(aq1, bk, s1);
    }

    const unsigned char mv = mb[k0 + lane];
    const unsigned long long bm = __ballot(mv != 0);
    const bool v0 = (bm >> lc) & 1ull;
    const bool v1 = (bm >> (16 + lc)) & 1ull;

    // ---- online softmax (rows replicated across 16-lane halves)
#pragma unroll
    for (int j = 0; j < 8; ++j) {
      float l0 = v0 ? s0[j] * SCALE : -1.0e30f;
      float l1 = v1 ? s1[j] * SCALE : -1.0e30f;
      float tm = fmaxf(l0, l1);
#pragma unroll
      for (int off = 8; off; off >>= 1) tm = fmaxf(tm, __shfl_xor(tm, off, 32));
      const float mnew = fmaxf(mrow[j], tm);
      const float p0 = __expf(l0 - mnew);
      const float p1 = __expf(l1 - mnew);
      float rs = p0 + p1;
#pragma unroll
      for (int off = 8; off; off >>= 1) rs += __shfl_xor(rs, off, 32);
      const float alpha = __expf(mrow[j] - mnew);
      lrow[j] = lrow[j] * alpha + rs;
      mrow[j] = mnew;
#pragma unroll
      for (int t = 0; t < 4; ++t) accv[t][j] *= alpha;
      const int row = j + 8 * lh;
      L.p[row * 40 + lc]      = (_Float16)p0;
      L.p[row * 40 + 16 + lc] = (_Float16)p1;
    }
    __builtin_amdgcn_wave_barrier();

    // ---- reload P as an A-fragment (16x32)
    v16h ap;
    {
      const _Float16* rowp = L.p + lc * 40;
      v8h c0 = *(const v8h*)(rowp + 16 * lh);
      v8h c1 = *(const v8h*)(rowp + 16 * lh + 8);
#pragma unroll
      for (int i = 0; i < 8; ++i) { ap[i] = c0[i]; ap[8 + i] = c1[i]; }
    }
    __builtin_amdgcn_wave_barrier();

    // ---- acc += P @ V
#pragma unroll
    for (int t = 0; t < 4; ++t) {
      v16h bv = lds_bfrag(L.vt[cur], VT_STRIDE, 16 * t, 0);
      accv[t] = WMMA_F16(ap, bv, accv[t]);
    }
  }

  // ---- finalize, store vals[b*S+s, head*64 + d] as f16
#pragma unroll
  for (int j = 0; j < 8; ++j) {
    const float rinv = (lrow[j] > 0.0f) ? (1.0f / lrow[j]) : 0.0f;
    const int s = q0 + j + 8 * lh;
    const size_t rowbase = ((size_t)bb * Sc + s) * Hc + head * DHc;
#pragma unroll
    for (int t = 0; t < 4; ++t) {
      vals[rowbase + 16 * t + lc] = (_Float16)(accv[t][j] * rinv);
    }
  }
}

// ---------------------------------------------------------------------------
// Kernel 3: output projection (pipelined), f32 out + bias.
// ---------------------------------------------------------------------------
__global__ void mha_out_gemm(const _Float16* __restrict__ vals,
                             const _Float16* __restrict__ ow,
                             const float*    __restrict__ o_b,
                             float* __restrict__ out) {
  const int wave = threadIdx.x >> 5;
  const int lane = threadIdx.x & 31;
  const int lh = lane >> 4, lc = lane & 15;
  const int m0 = blockIdx.x * 16;
  const int n0 = (blockIdx.y * 4 + wave) * 64;

  const _Float16* arow = vals + (size_t)m0 * Hc;

  v8f acc[4];
#pragma unroll
  for (int t = 0; t < 4; ++t) acc[t] = (v8f){};

  v16h a = load_a_frag(arow, Hc);
  v16h b[4];
#pragma unroll
  for (int t = 0; t < 4; ++t) b[t] = load_bt_frag(ow + (size_t)(n0 + 16 * t) * Hc, Hc);

  for (int kk = 0; kk < Hc; kk += 32) {
    v16h an = a, bn0 = b[0], bn1 = b[1], bn2 = b[2], bn3 = b[3];
    if (kk + 32 < Hc) {
      __builtin_prefetch(arow + kk + 64, 0, 1);
      an  = load_a_frag(arow + kk + 32, Hc);
      bn0 = load_bt_frag(ow + (size_t)(n0 +  0) * Hc + kk + 32, Hc);
      bn1 = load_bt_frag(ow + (size_t)(n0 + 16) * Hc + kk + 32, Hc);
      bn2 = load_bt_frag(ow + (size_t)(n0 + 32) * Hc + kk + 32, Hc);
      bn3 = load_bt_frag(ow + (size_t)(n0 + 48) * Hc + kk + 32, Hc);
    }
    acc[0] = WMMA_F16(a, b[0], acc[0]);
    acc[1] = WMMA_F16(a, b[1], acc[1]);
    acc[2] = WMMA_F16(a, b[2], acc[2]);
    acc[3] = WMMA_F16(a, b[3], acc[3]);
    a = an; b[0] = bn0; b[1] = bn1; b[2] = bn2; b[3] = bn3;
  }

#pragma unroll
  for (int t = 0; t < 4; ++t) {
    const int n = n0 + 16 * t + lc;
    const float bias = o_b[n];
#pragma unroll
    for (int j = 0; j < 8; ++j) {
      const int m = m0 + j + 8 * lh;
      out[(size_t)m * Hc + n] = acc[t][j] + bias;
    }
  }
}

// ---------------------------------------------------------------------------
extern "C" void kernel_launch(void* const* d_in, const int* in_sizes, int n_in,
                              void* d_out, int out_size, void* d_ws, size_t ws_size,
                              hipStream_t stream) {
  const float* x      = (const float*)d_in[0];                 // [4,2048,512]
  const unsigned char* mask = (const unsigned char*)d_in[1];   // [4,2048] bool
  const float* qkv_w  = (const float*)d_in[2];                 // [1536,512]
  const float* qkv_b  = (const float*)d_in[3];                 // [1536]
  const float* o_w    = (const float*)d_in[4];                 // [512,512]
  const float* o_b    = (const float*)d_in[5];                 // [512]
  float* out          = (float*)d_out;                         // [4,2048,512]

  _Float16* ws   = (_Float16*)d_ws;
  _Float16* xf   = ws + OFF_XF16;
  _Float16* wq   = ws + OFF_WQKV;
  _Float16* ow   = ws + OFF_OW;
  _Float16* Q    = ws + OFF_Q;
  _Float16* K    = ws + OFF_K;
  _Float16* Vt   = ws + OFF_VT;
  _Float16* vals = ws + OFF_VALS;

  {
    const int nx = Mrows * Hc;
    const int nw = 3 * Hc * Hc;
    const int no = Hc * Hc;
    mha_cvt_f16<<<(nx + 255) / 256, 256, 0, stream>>>(x, xf, nx);
    mha_cvt_f16<<<(nw + 255) / 256, 256, 0, stream>>>(qkv_w, wq, nw);
    mha_cvt_f16<<<(no + 255) / 256, 256, 0, stream>>>(o_w, ow, no);
  }

  mha_qkv_gemm<<<dim3(Mrows / 16, 6, 1), 128, 0, stream>>>(xf, wq, qkv_b, Q, K, Vt);
  mha_flash_attn<<<2048, 64, 0, stream>>>(Q, K, Vt, mask, vals);
  mha_out_gemm<<<dim3(Mrows / 16, 2, 1), 128, 0, stream>>>(vals, ow, o_b, out);
}
